// Net_58428735095641
// MI455X (gfx1250) — matrix operand order repaired and checked by reading, not compiled, not run on hardware
//
#include <hip/hip_runtime.h>

#define N_ROWS   12960      // 4 * 3240 pair-rows
#define HIDDEN   256
#define NPAIR    3240
#define NVAR     81
#define OOUT     81
#define RED_BLOCKS 81
#define RED_ROWS   160      // 81 * 160 = 12960

typedef __attribute__((ext_vector_type(16))) _Float16 v16h;
typedef __attribute__((ext_vector_type(8)))  _Float16 v8h;
typedef __attribute__((ext_vector_type(8)))  float    v8f;

// ---------------------------------------------------------------- pair table
__global__ void pairs_kernel(int* pi, int* pj) {
  int p = blockIdx.x * blockDim.x + threadIdx.x;
  if (p >= NPAIR) return;
  int i = 0, rem = p;
  while (rem >= (NVAR - 1) - i) { rem -= (NVAR - 1) - i; ++i; }
  pi[p] = i;
  pj[p] = i + 1 + rem;
}

// ------------------------------------------------- fp32 -> f16 weight images
__global__ void prep_weights_kernel(const float* __restrict__ w1,
                                    const float* __restrict__ w2,
                                    const float* __restrict__ wout,
                                    _Float16* __restrict__ w1h,
                                    _Float16* __restrict__ w2h,
                                    _Float16* __restrict__ wouth) {
  int idx = blockIdx.x * blockDim.x + threadIdx.x;
  const int NW = 2 * HIDDEN * HIDDEN;
  if (idx < NW) { w1h[idx] = (_Float16)w1[idx]; return; }
  idx -= NW;
  if (idx < NW) { w2h[idx] = (_Float16)w2[idx]; return; }
  idx -= NW;
  if (idx < 96 * HIDDEN) {            // pad 81 rows -> 96 (6 full o-tiles)
    int r = idx / HIDDEN, c = idx % HIDDEN;
    wouth[idx] = (r < OOUT) ? (_Float16)wout[r * HIDDEN + c] : (_Float16)0.0f;
  }
}

// ------------------------------------------- pair gather + input FC (K = 4)
__global__ void fcin_kernel(const float* __restrict__ x,
                            const float* __restrict__ w_in,
                            const float* __restrict__ b_in,
                            const int* __restrict__ pi, const int* __restrict__ pj,
                            float* __restrict__ h) {
  int idx = blockIdx.x * blockDim.x + threadIdx.x;   // exactly N_ROWS*HIDDEN
  int n = idx >> 8, o = idx & 255;
  int b = n / NPAIR, p = n % NPAIR;
  const float* xb = x + b * NVAR * 2;
  int i = pi[p], j = pj[p];
  const float* w = w_in + o * 4;
  float v = b_in[o] + xb[i*2] * w[0] + xb[i*2+1] * w[1]
                    + xb[j*2] * w[2] + xb[j*2+1] * w[3];
  h[idx] = fmaxf(v, 0.0f);
}

// ------------------------------------- BN stage 1: per-column partial sums
__global__ void colreduce_kernel(const float* __restrict__ src, float* __restrict__ partial) {
  int t = threadIdx.x;                 // column 0..255 (coalesced across threads)
  int b = blockIdx.x;
  const float* s = src + (size_t)b * RED_ROWS * HIDDEN + t;
  float sum = 0.f, sq = 0.f;
  for (int r = 0; r < RED_ROWS; ++r) {
    float v = s[(size_t)r * HIDDEN];
    sum += v; sq += v * v;
  }
  partial[b * 2 * HIDDEN + t]          = sum;
  partial[b * 2 * HIDDEN + HIDDEN + t] = sq;
}

// ------------------------------------- BN stage 2: fold into scale / shift
__global__ void bnfinalize_kernel(const float* __restrict__ partial,
                                  const float* __restrict__ g,
                                  const float* __restrict__ beta,
                                  float* __restrict__ scsh) {
  int t = threadIdx.x;
  float sum = 0.f, sq = 0.f;
  for (int b = 0; b < RED_BLOCKS; ++b) {
    sum += partial[b * 2 * HIDDEN + t];
    sq  += partial[b * 2 * HIDDEN + HIDDEN + t];
  }
  float mean = sum / (float)N_ROWS;
  float var  = sq  / (float)N_ROWS - mean * mean;   // biased, matches jnp.var
  float rstd = rsqrtf(var + 1e-5f);
  float sc = g[t] * rstd;
  scsh[t]          = sc;
  scsh[HIDDEN + t] = beta[t] - mean * sc;
}

// ----------------------------- apply BN + ReLU, cast GEMM operand to f16
__global__ void bnrelucast_kernel(const float* __restrict__ src,
                                  const float* __restrict__ scsh,
                                  _Float16* __restrict__ a16) {
  int idx = blockIdx.x * blockDim.x + threadIdx.x;  // exactly N_ROWS*HIDDEN
  int c = idx & 255;
  float v = src[idx] * scsh[c] + scsh[HIDDEN + c];
  a16[idx] = (_Float16)fmaxf(v, 0.0f);
}

// ------------------------------------------------ WMMA fragment loaders
// A (16x32 f16): lane holds row (lane&15); elems 0..7 = K 8*(lane>>4)+0..7,
//                elems 8..15 = K 16+8*(lane>>4)+0..7
__device__ __forceinline__ v16h load_a_frag(const _Float16* p) {
  v8h lo = *(const v8h*)(p);
  v8h hi = *(const v8h*)(p + 16);
  return __builtin_shufflevector(lo, hi, 0,1,2,3,4,5,6,7,8,9,10,11,12,13,14,15);
}
// B (32x16 f16): lane holds col (lane&15); elems 0..7 = K 8*(lane>>4)+0..7,
//                elems 8..15 = K 16+8*(lane>>4)+0..7
__device__ __forceinline__ v16h load_b_frag(const _Float16* p) {
  v8h lo = *(const v8h*)(p);
  v8h hi = *(const v8h*)(p + 16);
  return __builtin_shufflevector(lo, hi, 0,1,2,3,4,5,6,7,8,9,10,11,12,13,14,15);
}

// --------------------------------------------- WMMA GEMM: dst = A @ W^T (+res)
// A: N_ROWS x 256 f16 row-major; W: O x 256 f16 row-major; acc fp32.
// One wave owns a 32x32 D tile (2 m-tiles x 2 o-tiles): per K-step,
// 2 A-frags + 2 B-frags feed 4 independent WMMAs (frag:wmma = 1:1).
__global__ __launch_bounds__(256) void wmma_gemm_kernel(
    const _Float16* __restrict__ A, const _Float16* __restrict__ W,
    const float* __restrict__ bias, const float* __restrict__ res,
    float* __restrict__ dst, int O, int otiles) {
  int lane   = threadIdx.x & 31;
  int wave   = threadIdx.x >> 5;
  int otile0 = (blockIdx.y * 8 + wave) * 2;   // wave covers otile0, otile0+1
  if (otile0 >= otiles) return;               // wave-uniform
  int mtile0 = blockIdx.x * 2;                // wave covers mtile0, mtile0+1
  int nl = lane & 15;                         // A row-in-tile / B col-in-tile
  int kh = lane >> 4;                         // K half selector

  const _Float16* arow0 = A + ((size_t)(mtile0 * 16 + nl)) * HIDDEN + kh * 8;
  const _Float16* arow1 = arow0 + (size_t)16 * HIDDEN;
  const _Float16* brow0 = W + ((size_t)(otile0 * 16 + nl)) * HIDDEN + kh * 8;
  const _Float16* brow1 = brow0 + (size_t)16 * HIDDEN;

  v8f acc00 = {0.f,0.f,0.f,0.f,0.f,0.f,0.f,0.f};
  v8f acc01 = acc00, acc10 = acc00, acc11 = acc00;

  #pragma unroll
  for (int k0 = 0; k0 < HIDDEN; k0 += 32) {
    v16h a0 = load_a_frag(arow0 + k0);
    v16h a1 = load_a_frag(arow1 + k0);
    v16h b0 = load_b_frag(brow0 + k0);
    v16h b1 = load_b_frag(brow1 + k0);
    acc00 = __builtin_amdgcn_wmma_f32_16x16x32_f16(false, a0, false, b0, (short)0, acc00, false, false);
    acc01 = __builtin_amdgcn_wmma_f32_16x16x32_f16(false, a0, false, b1, (short)0, acc01, false, false);
    acc10 = __builtin_amdgcn_wmma_f32_16x16x32_f16(false, a1, false, b0, (short)0, acc10, false, false);
    acc11 = __builtin_amdgcn_wmma_f32_16x16x32_f16(false, a1, false, b1, (short)0, acc11, false, false);
  }

  // C/D layout: lane holds col (lane&15); VGPR v holds row 8*(lane>>4)+v.
  auto store_tile = [&](const v8f& acc, int mbase, int ocol) {
    if (ocol >= O) return;            // tail guard for O = 81
    float bv = bias[ocol];
    #pragma unroll
    for (int v = 0; v < 8; ++v) {
      size_t di = (size_t)(mbase + v) * O + ocol;
      float r = acc[v] + bv;
      if (res) r += res[di];          // residual stream (stride O == 256 here)
      dst[di] = r;
    }
  };
  int mb0 = mtile0 * 16 + kh * 8;
  int mb1 = mb0 + 16;
  int oc0 = otile0 * 16 + nl;
  int oc1 = oc0 + 16;
  store_tile(acc00, mb0, oc0);
  store_tile(acc01, mb0, oc1);
  store_tile(acc10, mb1, oc0);
  store_tile(acc11, mb1, oc1);
}

// ------------------------------------------- symmetric scatter to (4,81,81,9,9)
__global__ void scatter_kernel(const float* __restrict__ pred, float* __restrict__ out,
                               int total) {
  int idx = blockIdx.x * blockDim.x + threadIdx.x;
  if (idx >= total) return;
  int c = idx % 81;                   // gi*9 + gj
  int t = idx / 81;
  int j = t % NVAR; t /= NVAR;
  int i = t % NVAR;
  int b = t / NVAR;
  float val = 0.0f;
  if (i != j) {
    int gi = c / 9, gj = c % 9;
    int lo = i < j ? i : j, hi = i < j ? j : i;
    int p  = lo * (NVAR - 1) - lo * (lo - 1) / 2 + (hi - lo - 1);   // triu index
    int cc = (i < j) ? (gi * 9 + gj) : (gj * 9 + gi);               // transpose below diag
    val = pred[((size_t)b * NPAIR + p) * OOUT + cc];
  }
  out[idx] = val;
}

// =============================================================== launcher
extern "C" void kernel_launch(void* const* d_in, const int* in_sizes, int n_in,
                              void* d_out, int out_size, void* d_ws, size_t ws_size,
                              hipStream_t stream) {
  (void)in_sizes; (void)n_in; (void)out_size; (void)ws_size;
  const float* x     = (const float*)d_in[0];
  const float* w_in  = (const float*)d_in[1];
  const float* b_in  = (const float*)d_in[2];
  const float* g1    = (const float*)d_in[3];
  const float* beta1 = (const float*)d_in[4];
  const float* w1    = (const float*)d_in[5];
  const float* bias1 = (const float*)d_in[6];
  const float* g2    = (const float*)d_in[7];
  const float* beta2 = (const float*)d_in[8];
  const float* w2    = (const float*)d_in[9];
  const float* bias2 = (const float*)d_in[10];
  const float* fg    = (const float*)d_in[11];
  const float* fbeta = (const float*)d_in[12];
  const float* w_out = (const float*)d_in[13];
  const float* b_out = (const float*)d_in[14];
  float* out = (float*)d_out;

  // workspace carve-up (deterministic, ~38 MB)
  char* ws = (char*)d_ws;
  auto carve = [&](size_t bytes) { char* p = ws; ws += (bytes + 255) & ~(size_t)255; return p; };
  float*    h       = (float*)   carve((size_t)N_ROWS * HIDDEN * 4);
  float*    z       = (float*)   carve((size_t)N_ROWS * HIDDEN * 4);
  float*    pred    = (float*)   carve((size_t)N_ROWS * OOUT * 4);
  _Float16* a16     = (_Float16*)carve((size_t)N_ROWS * HIDDEN * 2);
  _Float16* w1h     = (_Float16*)carve((size_t)2 * HIDDEN * HIDDEN * 2);
  _Float16* w2h     = (_Float16*)carve((size_t)2 * HIDDEN * HIDDEN * 2);
  _Float16* wouth   = (_Float16*)carve((size_t)96 * HIDDEN * 2);
  float*    partial = (float*)   carve((size_t)RED_BLOCKS * 2 * HIDDEN * 4);
  float*    scsh    = (float*)   carve((size_t)2 * HIDDEN * 4);
  int*      pi      = (int*)     carve((size_t)NPAIR * 4);
  int*      pj      = (int*)     carve((size_t)NPAIR * 4);

  pairs_kernel<<<(NPAIR + 255) / 256, 256, 0, stream>>>(pi, pj);
  int prep_total = 2 * (2 * HIDDEN * HIDDEN) + 96 * HIDDEN;
  prep_weights_kernel<<<(prep_total + 255) / 256, 256, 0, stream>>>(w1, w2, w_out,
                                                                    w1h, w2h, wouth);
  fcin_kernel<<<N_ROWS * HIDDEN / 256, 256, 0, stream>>>(x, w_in, b_in, pi, pj, h);

  auto bn_gemm = [&](const float* src, const float* g, const float* beta,
                     const _Float16* Wh, const float* bias, const float* res,
                     float* dst, int O, int otiles) {
    colreduce_kernel<<<RED_BLOCKS, 256, 0, stream>>>(src, partial);
    bnfinalize_kernel<<<1, 256, 0, stream>>>(partial, g, beta, scsh);
    bnrelucast_kernel<<<N_ROWS * HIDDEN / 256, 256, 0, stream>>>(src, scsh, a16);
    // 32x32 D tile per wave: grid.x = N_ROWS/32, one block covers 16 o-tiles
    wmma_gemm_kernel<<<dim3(N_ROWS / 32, 1), 256, 0, stream>>>(a16, Wh, bias,
                                                               res, dst, O, otiles);
  };

  // ResBlock 0
  bn_gemm(h, g1,           beta1,           w1h,                 bias1,          nullptr, z, HIDDEN, 16);
  bn_gemm(z, g2,           beta2,           w2h,                 bias2,          h,       h, HIDDEN, 16);
  // ResBlock 1
  bn_gemm(h, g1 + HIDDEN,  beta1 + HIDDEN,  w1h + HIDDEN*HIDDEN, bias1 + HIDDEN, nullptr, z, HIDDEN, 16);
  bn_gemm(z, g2 + HIDDEN,  beta2 + HIDDEN,  w2h + HIDDEN*HIDDEN, bias2 + HIDDEN, h,       h, HIDDEN, 16);
  // final BN + output projection (O = 81, padded to 6 o-tiles)
  bn_gemm(h, fg, fbeta, wouth, b_out, nullptr, pred, OOUT, 6);

  int total = 4 * NVAR * NVAR * 81;
  scatter_kernel<<<(total + 255) / 256, 256, 0, stream>>>(pred, out, total);
}
